// InterpretableMultiHeadAttention_61744449847988
// MI455X (gfx1250) — compile-verified, matrix operand order checked by reading, not run on hardware
//
#include <hip/hip_runtime.h>

// Problem constants (fixed by the reference)
#define B_    8
#define S_    1024
#define H_    1024
#define NH_   16
#define D_    64
#define V_    1024
#define ROWS_ (B_ * S_)   // 8192

typedef __attribute__((ext_vector_type(16))) _Float16 v16h;
typedef __attribute__((ext_vector_type(8)))  _Float16 v8h;
typedef __attribute__((ext_vector_type(8)))  float    v8f;
typedef unsigned __attribute__((ext_vector_type(4))) u32x4;
typedef int      __attribute__((ext_vector_type(4))) i32x4;
typedef int      __attribute__((ext_vector_type(8))) i32x8;

__device__ __forceinline__ v16h cat16(v8h lo, v8h hi) {
  v16h r;
#pragma unroll
  for (int i = 0; i < 8; ++i) { r[i] = lo[i]; r[i + 8] = hi[i]; }
  return r;
}

__device__ __forceinline__ v8f wmma_f16(v16h a, v16h b, v8f c) {
  // D = A(16x32 f16) * B(32x16 f16) + C(16x16 f32)
  return __builtin_amdgcn_wmma_f32_16x16x32_f16(false, a, false, b, (short)0, c,
                                                false, false);
}

__device__ __forceinline__ void pack4(v16h& a, int base, float4 x) {
  a[base + 0] = (_Float16)x.x; a[base + 1] = (_Float16)x.y;
  a[base + 2] = (_Float16)x.z; a[base + 3] = (_Float16)x.w;
}

// ---------------------------------------------------------------------------
// Tensor Data Mover: 2-D tile (f16 elements) global -> LDS.
// Descriptor per CDNA5 ISA ch.8: group0 = {flags, lds_addr, global_addr, type},
// group1 = dims/strides. 2-D => groups 2/3 zero. All dwords forced uniform.
// ---------------------------------------------------------------------------
#if __has_builtin(__builtin_amdgcn_tensor_load_to_lds)
#define HAVE_TDM 1
__device__ __forceinline__ unsigned rfl(unsigned x) {
  return (unsigned)__builtin_amdgcn_readfirstlane((int)x);
}
__device__ __forceinline__ void tdm_load_2d_f16(unsigned lds_byte_addr,
                                                const void* gaddr,
                                                unsigned tile_w,   // elements
                                                unsigned tile_h,   // rows
                                                unsigned row_stride) { // elements
  const unsigned long long ga = (unsigned long long)gaddr;
  u32x4 g0;
  g0.x = 1u;                                        // count=1, user descriptor
  g0.y = rfl(lds_byte_addr);
  g0.z = rfl((unsigned)ga);
  g0.w = rfl((unsigned)((ga >> 32) & 0x01FFFFFFu) | 0x80000000u); // type=2
  i32x8 g1;
  g1[0] = (int)(1u << 16);                          // data_size=1 (2B), no pad/iter/mcast
  g1[1] = (int)rfl((row_stride & 0xFFFFu) << 16);   // tensor_dim0[15:0]
  g1[2] = (int)rfl(((row_stride >> 16) & 0xFFFFu) | ((tile_h & 0xFFFFu) << 16)); // dim0 hi | tensor_dim1 lo
  g1[3] = (int)rfl((tile_w & 0xFFFFu) << 16);       // tensor_dim1 hi (0) | tile_dim0
  g1[4] = (int)rfl(tile_h & 0xFFFFu);               // tile_dim1 | tile_dim2=0
  g1[5] = (int)rfl(row_stride);                     // tensor_dim0_stride[31:0]
  g1[6] = 0;
  g1[7] = 0;
  const i32x4 z4 = (i32x4)0;
#if defined(__clang_major__) && (__clang_major__ >= 23)
  const i32x8 z8 = (i32x8)0;
  __builtin_amdgcn_tensor_load_to_lds(g0, g1, z4, z4, z8, 0);
#else
  __builtin_amdgcn_tensor_load_to_lds(g0, g1, z4, z4, 0);
#endif
}
#else
#define HAVE_TDM 0
#endif

// ---------------------------------------------------------------------------
// Kernel 0: one-time weight transpose + f32->f16 convert (bandwidth-trivial).
// WqT/WkT: [n][d][h], WvT: [v][h]  => B-fragments become contiguous 16B loads.
// ---------------------------------------------------------------------------
__global__ void prep_weights_kernel(const float* __restrict__ Wq, const float* __restrict__ Wk,
                                    const float* __restrict__ Wv,
                                    _Float16* __restrict__ WqT, _Float16* __restrict__ WkT,
                                    _Float16* __restrict__ WvT) {
  const int idx = blockIdx.x * 256 + threadIdx.x;
  const int NQK = NH_ * D_ * H_;                    // 1 Mi elements each
  if (idx < 2 * NQK) {
    const int sel = idx >= NQK;
    const int r   = idx - sel * NQK;
    const int n   = r / (D_ * H_);
    const int rem = r % (D_ * H_);
    const int d   = rem / H_;
    const int h   = rem % H_;
    const float* W = sel ? Wk : Wq;
    _Float16*    o = sel ? WkT : WqT;
    o[r] = (_Float16)W[((size_t)n * H_ + h) * D_ + d];
  } else {
    const int r  = idx - 2 * NQK;                   // over V_*H_
    const int vc = r / H_;
    const int h  = r % H_;
    WvT[r] = (_Float16)Wv[(size_t)h * V_ + vc];
  }
}

// ---------------------------------------------------------------------------
// Kernel 1: per-head Q/K projections, one wave per 16(row) x 64(D) f16 tile.
// qh/kh layout: [n][row][d] row-major f16.
// ---------------------------------------------------------------------------
__global__ void proj_qk_kernel(const float* __restrict__ q,   const float* __restrict__ kin,
                               const _Float16* __restrict__ WqT, const float* __restrict__ bq,
                               const _Float16* __restrict__ WkT, const float* __restrict__ bk,
                               _Float16* __restrict__ qh, _Float16* __restrict__ kh) {
  const int tid  = blockIdx.x * blockDim.x + threadIdx.x;
  const int wid  = tid >> 5;
  const int lane = threadIdx.x & 31;
  const int sel  = wid >> 13;          // 0 = Q, 1 = K
  const int rem  = wid & 8191;
  const int n    = rem >> 9;           // head
  const int rt   = rem & 511;          // 16-row tile over the 8192 rows

  const float* X       = sel ? kin : q;
  const _Float16* WT   = (sel ? WkT : WqT) + (size_t)n * D_ * H_;
  const float* bb      = (sel ? bk : bq) + n * D_;
  _Float16* outp       = (sel ? kh : qh) + (size_t)n * ROWS_ * D_;

  const int l15 = lane & 15;
  const int kb8 = (lane >> 4) << 3;    // 0 or 8: lane-group K offset == C-row offset
  const int row = rt * 16 + l15;       // A-matrix row held by this lane
  const float* xrow = X + (size_t)row * H_;

  v8f acc[4] = {};
#pragma unroll 1
  for (int kk = 0; kk < H_ / 32; ++kk) {
    const int h0 = kk * 32 + kb8;
    v16h a;
    const float4* p0 = reinterpret_cast<const float4*>(xrow + h0);
    const float4* p1 = reinterpret_cast<const float4*>(xrow + h0 + 16);
    pack4(a, 0,  p0[0]); pack4(a, 4,  p0[1]);
    pack4(a, 8,  p1[0]); pack4(a, 12, p1[1]);
#pragma unroll
    for (int c = 0; c < 4; ++c) {
      const _Float16* wrow = WT + (size_t)(c * 16 + l15) * H_;   // B column per lane
      const v16h bfr = cat16(*(const v8h*)(wrow + h0), *(const v8h*)(wrow + h0 + 16));
      acc[c] = wmma_f16(a, bfr, acc[c]);
    }
  }
#pragma unroll
  for (int c = 0; c < 4; ++c) {
    const int d = c * 16 + l15;
    const float bias = bb[d];
#pragma unroll
    for (int r = 0; r < 8; ++r) {
      const int orow = rt * 16 + r + kb8;          // C layout: M = r + (lane/16)*8
      outp[(size_t)orow * D_ + d] = (_Float16)(acc[c][r] + bias);
    }
  }
}

// ---------------------------------------------------------------------------
// Kernel 2: shared V projection, stored TRANSPOSED: vvT[v][row] f16.
// ---------------------------------------------------------------------------
__global__ void proj_v_kernel(const float* __restrict__ v, const _Float16* __restrict__ WvT,
                              const float* __restrict__ bv, _Float16* __restrict__ vvT) {
  const int tid  = blockIdx.x * blockDim.x + threadIdx.x;
  const int wid  = tid >> 5;
  const int lane = threadIdx.x & 31;
  const int rt   = wid >> 4;           // 16-row tile
  const int wt   = wid & 15;           // 64-col group of V
  const int l15  = lane & 15;
  const int kb8  = (lane >> 4) << 3;
  const int row  = rt * 16 + l15;
  const float* xrow = v + (size_t)row * H_;

  v8f acc[4] = {};
#pragma unroll 1
  for (int kk = 0; kk < H_ / 32; ++kk) {
    const int h0 = kk * 32 + kb8;
    v16h a;
    const float4* p0 = reinterpret_cast<const float4*>(xrow + h0);
    const float4* p1 = reinterpret_cast<const float4*>(xrow + h0 + 16);
    pack4(a, 0,  p0[0]); pack4(a, 4,  p0[1]);
    pack4(a, 8,  p1[0]); pack4(a, 12, p1[1]);
#pragma unroll
    for (int c = 0; c < 4; ++c) {
      const _Float16* wrow = WvT + (size_t)(wt * 64 + c * 16 + l15) * H_;
      const v16h bfr = cat16(*(const v8h*)(wrow + h0), *(const v8h*)(wrow + h0 + 16));
      acc[c] = wmma_f16(a, bfr, acc[c]);
    }
  }
#pragma unroll
  for (int c = 0; c < 4; ++c) {
    const int d = wt * 64 + c * 16 + l15;
    const float bias = bv[d];
#pragma unroll
    for (int r = 0; r < 8; ++r) {
      const int orow = rt * 16 + r + kb8;
      vvT[(size_t)d * ROWS_ + orow] = (_Float16)(acc[c][r] + bias);
    }
  }
}

// ---------------------------------------------------------------------------
// Kernel 3: fused causal flash attention + head-mean.
// Block = 16 query rows of one batch, 8 waves; wave owns a 128-wide V stripe.
// K tiles (32x64 f16, shared by all 8 waves) are DMA'd into LDS by the TDM,
// double-buffered so the copy of block kb+32 overlaps the math on block kb.
// ---------------------------------------------------------------------------
__global__ __launch_bounds__(256) void attn_kernel(const _Float16* __restrict__ qh,
                                                   const _Float16* __restrict__ kh,
                                                   const _Float16* __restrict__ vvT,
                                                   float* __restrict__ out) {
  __shared__ alignas(16) _Float16 Ktile[2][32 * 64];   // 2 x 4 KB double buffer
  __shared__ alignas(16) _Float16 Pbuf[8][16 * 32];    // per-wave P tile (1 KB)

  const int b     = blockIdx.x >> 6;   // batch
  const int rt    = blockIdx.x & 63;   // 16-row tile within the sequence
  const int wave  = threadIdx.x >> 5;
  const int lane  = threadIdx.x & 31;
  const int l15   = lane & 15;
  const int kb8   = (lane >> 4) << 3;
  const int vbase = wave * 128;        // this wave's V-column stripe
  const int arow  = b * S_ + rt * 16 + l15;
  const int kvend = rt * 16 + 16;      // causal: keys t <= s   (uniform per block)

  v8f G[8] = {};                       // head-mean accumulator, 16 x 128
  const float scale = 0.125f;          // 64^-0.5

  for (int n = 0; n < NH_; ++n) {
    const _Float16* khead = kh + (size_t)n * ROWS_ + 0;   // indexed below
    // Q tile for this head: A fragments for K=0..31 and K=32..63
    const _Float16* qrow = qh + ((size_t)n * ROWS_ + arow) * D_;
    const v16h a0 = cat16(*(const v8h*)(qrow + kb8),      *(const v8h*)(qrow + kb8 + 16));
    const v16h a1 = cat16(*(const v8h*)(qrow + 32 + kb8), *(const v8h*)(qrow + 48 + kb8));

    v8f acc[8] = {};
    float m[8], lsum[8];
#pragma unroll
    for (int r = 0; r < 8; ++r) { m[r] = -1e30f; lsum[r] = 0.0f; }

    // ---- prologue: stage K tile for kb=0 into buffer 0 ----
    if (wave == 0) {
      const _Float16* src = kh + ((size_t)n * ROWS_ + b * S_) * D_;
#if HAVE_TDM
      tdm_load_2d_f16((unsigned)(size_t)(const void*)&Ktile[0][0], src, 64, 32, 64);
#else
      const v8h* s8 = (const v8h*)(src + (size_t)lane * 64);
      v8h* d8 = (v8h*)&Ktile[0][lane * 64];
#pragma unroll
      for (int j = 0; j < 8; ++j) d8[j] = s8[j];
#endif
    }

    int bufi = 0;
    for (int kb = 0; kb < kvend; kb += 32, bufi ^= 1) {
      const int nxt = bufi ^ 1;
      __syncthreads();                           // reads of Ktile[nxt] (iter-1) done
      if (wave == 0 && (kb + 32) < kvend) {
        const _Float16* src = kh + ((size_t)n * ROWS_ + b * S_ + kb + 32) * D_;
#if HAVE_TDM
        tdm_load_2d_f16((unsigned)(size_t)(const void*)&Ktile[nxt][0], src, 64, 32, 64);
#else
        const v8h* s8 = (const v8h*)(src + (size_t)lane * 64);
        v8h* d8 = (v8h*)&Ktile[nxt][lane * 64];
#pragma unroll
        for (int j = 0; j < 8; ++j) d8[j] = s8[j];
#endif
      }
#if HAVE_TDM
      if (wave == 0) {
        if ((kb + 32) < kvend) __builtin_amdgcn_s_wait_tensorcnt(1);
        else                   __builtin_amdgcn_s_wait_tensorcnt(0);
      }
#endif
      __syncthreads();                           // Ktile[bufi] ready for everyone

      // ---- scores: qh(16x64) @ K^T from LDS -> 16x32 (two 16x16 tiles) ----
      v8f sc[2] = {};
#pragma unroll
      for (int c = 0; c < 2; ++c) {
        const _Float16* krow = &Ktile[bufi][(c * 16 + l15) * 64]; // B col = key row
        const v16h b0 = cat16(*(const v8h*)(krow + kb8),      *(const v8h*)(krow + kb8 + 16));
        const v16h b1 = cat16(*(const v8h*)(krow + 32 + kb8), *(const v8h*)(krow + 48 + kb8));
        sc[c] = wmma_f16(a0, b0, sc[c]);
        sc[c] = wmma_f16(a1, b1, sc[c]);
      }
      // prefetch next V block while softmax runs
      if ((kb + 32) < kvend) {
        __builtin_prefetch(vvT + (size_t)(vbase + l15) * ROWS_ + b * S_ + kb + 32, 0, 0);
      }
      // ---- scale + causal mask ----
#pragma unroll
      for (int c = 0; c < 2; ++c) {
        const int t = kb + c * 16 + l15;
#pragma unroll
        for (int r = 0; r < 8; ++r) {
          const int srow = rt * 16 + r + kb8;
          const float sv = sc[c][r] * scale;
          sc[c][r] = (t <= srow) ? sv : -1e30f;
        }
      }
      // ---- online softmax: row max via half-wave shfl_xor ----
      float alpha[8];
#pragma unroll
      for (int r = 0; r < 8; ++r) {
        float vmx = fmaxf(sc[0][r], sc[1][r]);
        vmx = fmaxf(vmx, __shfl_xor(vmx, 1, 32));
        vmx = fmaxf(vmx, __shfl_xor(vmx, 2, 32));
        vmx = fmaxf(vmx, __shfl_xor(vmx, 4, 32));
        vmx = fmaxf(vmx, __shfl_xor(vmx, 8, 32));
        const float nm = fmaxf(m[r], vmx);
        alpha[r] = __expf(m[r] - nm);
        m[r] = nm;
      }
      // ---- P = exp(s - m): row sums + f16 spill to LDS for C->A transpose ----
      _Float16* P = &Pbuf[wave][0];
#pragma unroll
      for (int c = 0; c < 2; ++c)
#pragma unroll
        for (int r = 0; r < 8; ++r) {
          const float p = __expf(sc[c][r] - m[r]);
          sc[c][r] = p;
          P[(r + kb8) * 32 + c * 16 + l15] = (_Float16)p;
        }
#pragma unroll
      for (int r = 0; r < 8; ++r) {
        float vs = sc[0][r] + sc[1][r];
        vs += __shfl_xor(vs, 1, 32);
        vs += __shfl_xor(vs, 2, 32);
        vs += __shfl_xor(vs, 4, 32);
        vs += __shfl_xor(vs, 8, 32);
        lsum[r] = lsum[r] * alpha[r] + vs;
      }
#pragma unroll
      for (int t8 = 0; t8 < 8; ++t8)
#pragma unroll
        for (int r = 0; r < 8; ++r) acc[t8][r] *= alpha[r];

      // intra-wave cross-lane LDS handoff (C-layout -> A-layout)
      asm volatile("s_wait_dscnt 0" ::: "memory");
      const _Float16* Pr = P + l15 * 32;
      const v16h pa = cat16(*(const v8h*)(Pr + kb8), *(const v8h*)(Pr + kb8 + 16));

      // ---- acc += P(16x32) @ vv(32 x 128-stripe) ----
#pragma unroll
      for (int t8 = 0; t8 < 8; ++t8) {
        const int vcol = vbase + t8 * 16 + l15;
        const _Float16* vp = vvT + (size_t)vcol * ROWS_ + b * S_ + kb;
        const v16h vb = cat16(*(const v8h*)(vp + kb8), *(const v8h*)(vp + kb8 + 16));
        acc[t8] = wmma_f16(pa, vb, acc[t8]);
      }
    }
    // fold this head into the mean: G += acc / (lsum * NH)
#pragma unroll
    for (int r = 0; r < 8; ++r) {
      const float inv = 0.0625f / lsum[r];
#pragma unroll
      for (int t8 = 0; t8 < 8; ++t8) G[t8][r] += acc[t8][r] * inv;
    }
    (void)khead;
  }
  // ---- store (B,S,V) fp32 ----
#pragma unroll
  for (int t8 = 0; t8 < 8; ++t8)
#pragma unroll
    for (int r = 0; r < 8; ++r) {
      const int srow = rt * 16 + r + kb8;
      const int vcol = vbase + t8 * 16 + l15;
      out[((size_t)b * S_ + srow) * V_ + vcol] = G[t8][r];
    }
}

// ---------------------------------------------------------------------------
extern "C" void kernel_launch(void* const* d_in, const int* in_sizes, int n_in,
                              void* d_out, int out_size, void* d_ws, size_t ws_size,
                              hipStream_t stream) {
  (void)in_sizes; (void)n_in; (void)out_size; (void)ws_size;
  const float* q  = (const float*)d_in[0];
  const float* k  = (const float*)d_in[1];
  const float* v  = (const float*)d_in[2];
  const float* Wq = (const float*)d_in[3];
  const float* bq = (const float*)d_in[4];
  const float* Wk = (const float*)d_in[5];
  const float* bk = (const float*)d_in[6];
  const float* Wv = (const float*)d_in[7];
  const float* bv = (const float*)d_in[8];
  float* out = (float*)d_out;

  // workspace: qh | kh | vvT | WqT | WkT | WvT (all f16)
  _Float16* qh  = (_Float16*)d_ws;
  _Float16* kh  = qh  + (size_t)NH_ * ROWS_ * D_;     // +16 MB
  _Float16* vvT = kh  + (size_t)NH_ * ROWS_ * D_;     // +16 MB
  _Float16* WqT = vvT + (size_t)V_ * ROWS_;           // +16 MB
  _Float16* WkT = WqT + (size_t)NH_ * D_ * H_;        // +2 MB
  _Float16* WvT = WkT + (size_t)NH_ * D_ * H_;        // +2 MB

  // 2*NH*D*H + V*H = 3 Mi elements -> 12288 blocks of 256
  prep_weights_kernel<<<12288, 256, 0, stream>>>(Wq, Wk, Wv, WqT, WkT, WvT);
  // 2 * NH * (8192/16) = 16384 waves -> 2048 blocks of 8 waves
  proj_qk_kernel<<<2048, 256, 0, stream>>>(q, k, WqT, bq, WkT, bk, qh, kh);
  // (8192/16) * (1024/64) = 8192 waves -> 1024 blocks
  proj_v_kernel<<<1024, 256, 0, stream>>>(v, WvT, bv, vvT);
  // B * S/16 = 512 blocks, 8 waves each
  attn_kernel<<<512, 256, 0, stream>>>(qh, kh, vvT, out);
}
